// AttentionBlock_16398185136290
// MI455X (gfx1250) — matrix operand order repaired and checked by reading, not compiled, not run on hardware
//
#include <hip/hip_runtime.h>

#define BATCH  8
#define CDIM   512
#define LDIM   4096
#define NGROUP 8
#define CPG    (CDIM / NGROUP)   // 64 channels per group
#define EPSV   1e-5f
#define NKB    (LDIM / 64)       // 64 key blocks

typedef __attribute__((ext_vector_type(16))) _Float16 v16h;
typedef __attribute__((ext_vector_type(8)))  _Float16 v8h;
typedef __attribute__((ext_vector_type(8)))  float    v8f;
typedef __attribute__((ext_vector_type(4)))  float    v4f;
typedef __attribute__((ext_vector_type(4)))  unsigned int u32x4;
typedef __attribute__((ext_vector_type(4)))  int      i32x4;
typedef __attribute__((ext_vector_type(8)))  int      i32x8;

#if defined(__has_builtin)
#if __has_builtin(__builtin_amdgcn_tensor_load_to_lds) && \
    __has_builtin(__builtin_amdgcn_s_wait_tensorcnt)
#define HAVE_TDM 1
#endif
#endif
#ifndef HAVE_TDM
#define HAVE_TDM 0
#endif

union F16x16 { v16h v; v8h h[2]; };

// A-fragment (16x32 f16, M x K), source row-major [M][K], stride_h halves/row.
// lane<16 -> row M=lane, K 0..7 (v0-v3) + 16..23 (v4-v7); lane>=16 -> K 8..15 + 24..31.
__device__ __forceinline__ v16h load_a_frag(const _Float16* row0, int stride_h, int lane) {
  const _Float16* p = row0 + (lane & 15) * stride_h + ((lane < 16) ? 0 : 8);
  F16x16 f;
  f.h[0] = *(const v8h*)(p);
  f.h[1] = *(const v8h*)(p + 16);
  return f.v;
}

// B-fragment (32x16 f16, K x N), source [N][K], stride_h halves per N row.
// lane<16 -> column N=lane, K 0..7 + 16..23; lane>=16 -> K 8..15 + 24..31.
__device__ __forceinline__ v16h load_b_frag(const _Float16* n0, int stride_h, int lane) {
  const _Float16* p = n0 + (lane & 15) * stride_h + ((lane < 16) ? 0 : 8);
  F16x16 f;
  f.h[0] = *(const v8h*)(p);
  f.h[1] = *(const v8h*)(p + 16);
  return f.v;
}

__device__ __forceinline__ v8f wmma16(v16h a, v16h b, v8f c) {
  return __builtin_amdgcn_wmma_f32_16x16x32_f16(false, a, false, b, (short)0, c, false, false);
}

#if HAVE_TDM
// Issue a TDM 2D tile load (f16 elements): tile_x contiguous elements per row,
// tile_y rows, stride_el elements between rows; packed contiguously into LDS.
__device__ __forceinline__ void tdm_load_2d(unsigned int ldsoff, const _Float16* g,
                                            unsigned int tile_x, unsigned int tile_y,
                                            unsigned int stride_el) {
  unsigned long long ga = (unsigned long long)(size_t)g;
  u32x4 g0;
  g0[0] = 1u;                                   // count=1, user descriptor
  g0[1] = ldsoff;                               // lds_addr (bytes)
  g0[2] = (unsigned int)ga;                     // global_addr[31:0]
  g0[3] = (unsigned int)((ga >> 32) & 0x01FFFFFFu) | (2u << 30);  // addr[56:32] | type=2
  i32x8 g1;
  g1[0] = 1 << 16;                              // data_size = 2 bytes
  g1[1] = (int)((stride_el & 0xFFFFu) << 16);   // tensor_dim0[15:0]
  g1[2] = (int)((stride_el >> 16) & 0xFFFFu);   // tensor_dim0[31:16], tensor_dim1[15:0]=0
  g1[3] = (int)(1u | (tile_x << 16));           // tensor_dim1[31:16]=1 (dim1=65536), tile_dim0
  g1[4] = (int)tile_y;                          // tile_dim1, tile_dim2=0
  g1[5] = (int)stride_el;                       // tensor_dim0_stride[31:0]
  g1[6] = 0;
  g1[7] = 0;
  i32x4 gz = {0, 0, 0, 0};
#if __clang_major__ >= 23
  i32x8 gz8 = {0, 0, 0, 0, 0, 0, 0, 0};
  __builtin_amdgcn_tensor_load_to_lds(g0, g1, gz, gz, gz8, 0);
#else
  __builtin_amdgcn_tensor_load_to_lds(g0, g1, gz, gz, 0);
#endif
}
#endif

__device__ __forceinline__ unsigned int lds_off(const void* p) {
  return (unsigned int)(unsigned long long)(size_t)p;   // low 32 bits = LDS offset
}

// ---------------------------------------------------------------------------
// Kernel 1: convert weights to f16 once.
// ---------------------------------------------------------------------------
__global__ __launch_bounds__(256) void k_cvt_w(const float* __restrict__ wqkv,
                                               const float* __restrict__ wout,
                                               _Float16* __restrict__ wqkv16,
                                               _Float16* __restrict__ wout16) {
  int i = blockIdx.x * 256 + threadIdx.x;
  const int n1 = 3 * CDIM * CDIM;
  if (i < n1)           wqkv16[i] = (_Float16)wqkv[i];
  if (i < CDIM * CDIM)  wout16[i] = (_Float16)wout[i];
}

// ---------------------------------------------------------------------------
// Kernel 2: GroupNorm statistics -> per-(batch,channel) affine a*x + b.
// ---------------------------------------------------------------------------
__global__ __launch_bounds__(256) void k_gn_stats(const float* __restrict__ x,
                                                  const float* __restrict__ gamma,
                                                  const float* __restrict__ beta,
                                                  float* __restrict__ scaleA,
                                                  float* __restrict__ biasB) {
  __shared__ float s1[256], s2[256];
  const int tid = threadIdx.x;
  const int b = blockIdx.x >> 3;
  const int g = blockIdx.x & 7;
  const float* base = x + ((size_t)b * CDIM + g * CPG) * LDIM;
  const int nelem = CPG * LDIM;
  float s = 0.f, sq = 0.f;
  for (int i = tid * 4; i < nelem; i += 256 * 4) {
    v4f v = *(const v4f*)(base + i);
#pragma unroll
    for (int j = 0; j < 4; ++j) { s += v[j]; sq += v[j] * v[j]; }
  }
  s1[tid] = s; s2[tid] = sq;
  __syncthreads();
  for (int st = 128; st > 0; st >>= 1) {
    if (tid < st) { s1[tid] += s1[tid + st]; s2[tid] += s2[tid + st]; }
    __syncthreads();
  }
  const float mean = s1[0] / (float)nelem;
  const float var  = s2[0] / (float)nelem - mean * mean;
  const float rstd = rsqrtf(var + EPSV);
  if (tid < CPG) {
    const int c = g * CPG + tid;
    const float a = gamma[c] * rstd;
    scaleA[b * CDIM + c] = a;
    biasB[b * CDIM + c]  = beta[c] - mean * a;
  }
}

// ---------------------------------------------------------------------------
// Kernel 3: fused GroupNorm + QKV GEMM.  Block tile 128x128, K-step 32.
//   q -> qT[B][L][C], k -> kT[B][L][C], v -> vN[B][C][L]
// ---------------------------------------------------------------------------
__global__ __launch_bounds__(256) void k_gn_qkv_gemm(
    const float* __restrict__ x, const _Float16* __restrict__ wqkv16,
    const float* __restrict__ bqkv, const float* __restrict__ scaleA,
    const float* __restrict__ biasB, _Float16* __restrict__ qT,
    _Float16* __restrict__ kT, _Float16* __restrict__ vN) {
  __shared__ _Float16 Al[128 * 32];   // [m][k]
  __shared__ _Float16 Bl[128 * 32];   // [l][k]
  const int tid = threadIdx.x, lane = tid & 31, wid = tid >> 5;
  const int b = blockIdx.z;
  const int obase = blockIdx.y * 128;
  const int lbase = blockIdx.x * 128;
  const int wm = (wid >> 2) * 64;
  const int wn = (wid & 3) * 32;
  const float* xb = x + (size_t)b * CDIM * LDIM;
  const float* sA = scaleA + b * CDIM;
  const float* sB = biasB + b * CDIM;

  v8f acc[4][2];
#pragma unroll
  for (int m = 0; m < 4; ++m)
#pragma unroll
    for (int n = 0; n < 2; ++n) acc[m][n] = (v8f)0.f;

  for (int kk = 0; kk < CDIM; kk += 32) {
    {
      const int r = tid >> 1, ch = (tid & 1) * 16;
      *(v16h*)(Al + r * 32 + ch) =
          *(const v16h*)(wqkv16 + (size_t)(obase + r) * CDIM + kk + ch);
    }
    {
      const int c = kk + (tid >> 3);
      const int lch = (tid & 7) * 16;
      const float a = sA[c], bb = sB[c];
      const float* xp = xb + (size_t)c * LDIM + lbase + lch;
      if (kk + 32 < CDIM)
        __builtin_prefetch(xp + 32 * LDIM, 0, 1);
#pragma unroll
      for (int i = 0; i < 16; i += 4) {
        v4f xv = *(const v4f*)(xp + i);
#pragma unroll
        for (int j = 0; j < 4; ++j)
          Bl[(lch + i + j) * 32 + (c - kk)] = (_Float16)(xv[j] * a + bb);
      }
    }
    __syncthreads();
    v16h af[4];
#pragma unroll
    for (int mt = 0; mt < 4; ++mt)
      af[mt] = load_a_frag(Al + (wm + mt * 16) * 32, 32, lane);
#pragma unroll
    for (int nt = 0; nt < 2; ++nt) {
      v16h bf = load_b_frag(Bl + (wn + nt * 16) * 32, 32, lane);
#pragma unroll
      for (int mt = 0; mt < 4; ++mt) acc[mt][nt] = wmma16(af[mt], bf, acc[mt][nt]);
    }
    __syncthreads();
  }

  const int lo = lane & 15;
  const int moff = (lane < 16) ? 0 : 8;
#pragma unroll
  for (int mt = 0; mt < 4; ++mt) {
    const int o0 = obase + wm + mt * 16 + moff;
    const int sec = o0 >> 9;
    const int oc = o0 & (CDIM - 1);
#pragma unroll
    for (int nt = 0; nt < 2; ++nt) {
      const int l = lbase + wn + nt * 16 + lo;
      v8f t = acc[mt][nt];
      if (sec == 0) {
        v8h h;
#pragma unroll
        for (int i = 0; i < 8; ++i) h[i] = (_Float16)(t[i] + bqkv[o0 + i]);
        *(v8h*)(qT + ((size_t)b * LDIM + l) * CDIM + oc) = h;
      } else if (sec == 1) {
        v8h h;
#pragma unroll
        for (int i = 0; i < 8; ++i) h[i] = (_Float16)(t[i] + bqkv[o0 + i]);
        *(v8h*)(kT + ((size_t)b * LDIM + l) * CDIM + oc) = h;
      } else {
#pragma unroll
        for (int i = 0; i < 8; ++i)
          vN[((size_t)b * CDIM + oc + i) * LDIM + l] = (_Float16)(t[i] + bqkv[o0 + i]);
      }
    }
  }
}

// ---------------------------------------------------------------------------
// Kernel 4: flash attention.  One block per (64-query block, batch), 8 waves.
// K tiles double-buffered + V tiles staged in LDS via the Tensor Data Mover
// (wave 0 issues, s_wait_tensorcnt + barrier synchronize), prefetched one
// block ahead so the DMA overlaps S-WMMA and softmax phases.
// ---------------------------------------------------------------------------
__global__ __launch_bounds__(256) void k_flash(const _Float16* __restrict__ qT,
                                               const _Float16* __restrict__ kT,
                                               const _Float16* __restrict__ vN,
                                               _Float16* __restrict__ O) {
  extern __shared__ char smem[];
  _Float16* Qs = (_Float16*)smem;                              // [64][512]   64 KB
  _Float16* Ks = (_Float16*)(smem + 65536);                    // [2][64][512]128 KB
  _Float16* Vs = (_Float16*)(smem + 65536 + 131072);           // [512][64]   64 KB
  float*    Ss = (float*)(smem + 65536 + 131072 + 65536);      // [64][64]    16 KB
  _Float16* Ps = (_Float16*)((char*)Ss + 64 * 64 * 4);         // [64][64]     8 KB
  float* rowm = (float*)((char*)Ps + 64 * 64 * 2);
  float* rowl = rowm + 64;
  float* corr = rowl + 64;

  const int tid = threadIdx.x, lane = tid & 31, wid = tid >> 5;
  const int b = blockIdx.y;
  const int q0 = blockIdx.x * 64;
  const float scale = 0.04419417382415922f;   // 512^-0.5
  const int cb = wid * 64;
  const int moff = (lane < 16) ? 0 : 8;

  const _Float16* kTb = kT + (size_t)b * LDIM * CDIM;
  const _Float16* vNb = vN + (size_t)b * CDIM * LDIM;

  // Stage Q block (contiguous rows from qT).
  {
    const int r = tid >> 2, h0 = (tid & 3) * 128;
    const _Float16* src = qT + ((size_t)b * LDIM + q0 + r) * CDIM + h0;
#pragma unroll
    for (int i = 0; i < 128; i += 16)
      *(v16h*)(Qs + r * 512 + h0 + i) = *(const v16h*)(src + i);
  }
  if (tid < 64) { rowm[tid] = -3.0e38f; rowl[tid] = 0.f; }
  v8f oacc[4][4];
#pragma unroll
  for (int m = 0; m < 4; ++m)
#pragma unroll
    for (int n = 0; n < 4; ++n) oacc[m][n] = (v8f)0.f;

#if HAVE_TDM
  const unsigned int ks_off0 = lds_off(Ks);
  const unsigned int ks_off1 = lds_off(Ks + 64 * 512);
  const unsigned int vs_off  = lds_off(Vs);
  if (wid == 0) {
    tdm_load_2d(ks_off0, kTb, 512, 64, CDIM);      // K block 0: 64 key-rows x 512 c
    tdm_load_2d(vs_off,  vNb, 64, 512, LDIM);      // V block 0: 512 c-rows x 64 keys
  }
#endif
  __syncthreads();

  for (int ib = 0; ib < NKB; ++ib) {
    const int key0 = ib * 64;
    _Float16* kcur = Ks + (ib & 1) * (64 * 512);
#if HAVE_TDM
    if (wid == 0 && ib > 0)
      tdm_load_2d(vs_off, vNb + key0, 64, 512, LDIM);
    // K(ib) issued before V(ib); in-order completion => <=1 outstanding means K done.
    __builtin_amdgcn_s_wait_tensorcnt(1);
    __syncthreads();
#else
    {  // cooperative staging fallback
      const int r = tid >> 2, h0 = (tid & 3) * 128;
      const _Float16* src = kTb + (size_t)(key0 + r) * CDIM + h0;
#pragma unroll
      for (int i = 0; i < 128; i += 16)
        *(v16h*)(kcur + r * 512 + h0 + i) = *(const v16h*)(src + i);
      const int r2 = tid * 2;
#pragma unroll
      for (int rr = 0; rr < 2; ++rr) {
        const _Float16* sv = vNb + (size_t)(r2 + rr) * LDIM + key0;
#pragma unroll
        for (int i = 0; i < 64; i += 16)
          *(v16h*)(Vs + (r2 + rr) * 64 + i) = *(const v16h*)(sv + i);
      }
    }
    __syncthreads();
#endif
    // --- phase a: S tiles (wave handles tiles 2*wid, 2*wid+1) ---
    {
      const int t0 = wid * 2;
      const int mt = t0 >> 2, nt0 = t0 & 3, nt1 = nt0 + 1;
      v8f s0 = (v8f)0.f, s1 = (v8f)0.f;
#pragma unroll
      for (int ks = 0; ks < 16; ++ks) {
        v16h a  = load_a_frag(Qs + mt * 16 * 512 + ks * 32, 512, lane);
        v16h b0 = load_b_frag(kcur + nt0 * 16 * 512 + ks * 32, 512, lane);
        v16h b1 = load_b_frag(kcur + nt1 * 16 * 512 + ks * 32, 512, lane);
        s0 = wmma16(a, b0, s0);
        s1 = wmma16(a, b1, s1);
      }
      const int m0 = mt * 16 + moff, n = lane & 15;
#pragma unroll
      for (int i = 0; i < 8; ++i) {
        Ss[(m0 + i) * 64 + nt0 * 16 + n] = s0[i] * scale;
        Ss[(m0 + i) * 64 + nt1 * 16 + n] = s1[i] * scale;
      }
    }
#if HAVE_TDM
    if (wid == 0 && ib + 1 < NKB)
      tdm_load_2d(((ib + 1) & 1) ? ks_off1 : ks_off0,
                  kTb + (size_t)(key0 + 64) * CDIM, 512, 64, CDIM);
#endif
    __syncthreads();
    // --- phase b: online softmax (one thread per query row) ---
    if (tid < 64) {
      const float mprev = rowm[tid];
      float mx = mprev;
      const float* sr = Ss + tid * 64;
#pragma unroll 8
      for (int j = 0; j < 64; ++j) mx = fmaxf(mx, sr[j]);
      const float cr = __expf(mprev - mx);
      float sum = 0.f;
      _Float16* pr = Ps + tid * 64;
#pragma unroll 8
      for (int j = 0; j < 64; ++j) {
        const float p = __expf(sr[j] - mx);
        sum += p;
        pr[j] = (_Float16)p;
      }
      rowl[tid] = rowl[tid] * cr + sum;
      rowm[tid] = mx;
      corr[tid] = cr;
    }
#if HAVE_TDM
    // V(ib) done when at most K(ib+1) remains outstanding.
    if (ib + 1 < NKB) __builtin_amdgcn_s_wait_tensorcnt(1);
    else              __builtin_amdgcn_s_wait_tensorcnt(0);
#endif
    __syncthreads();
    // --- phase c: rescale O, accumulate O += P V^T ---
    {
#pragma unroll
      for (int mt = 0; mt < 4; ++mt) {
        v8f cv;
#pragma unroll
        for (int i = 0; i < 8; ++i) cv[i] = corr[mt * 16 + moff + i];
#pragma unroll
        for (int nt = 0; nt < 4; ++nt)
#pragma unroll
          for (int i = 0; i < 8; ++i) oacc[mt][nt][i] *= cv[i];
      }
#pragma unroll
      for (int ks = 0; ks < 2; ++ks)
#pragma unroll
        for (int nt = 0; nt < 4; ++nt) {
          v16h bf = load_b_frag(Vs + (cb + nt * 16) * 64 + ks * 32, 64, lane);
#pragma unroll
          for (int mt = 0; mt < 4; ++mt) {
            v16h af = load_a_frag(Ps + mt * 16 * 64 + ks * 32, 64, lane);
            oacc[mt][nt] = wmma16(af, bf, oacc[mt][nt]);
          }
        }
    }
    __syncthreads();
  }

  // Finalize: O / rowl -> O[B][C][L] (contiguous v8h per lane).
  {
    const int n = lane & 15;
#pragma unroll
    for (int mt = 0; mt < 4; ++mt) {
      v8f inv;
#pragma unroll
      for (int i = 0; i < 8; ++i) inv[i] = 1.0f / rowl[mt * 16 + moff + i];
#pragma unroll
      for (int nt = 0; nt < 4; ++nt) {
        const int c = cb + nt * 16 + n;
        v8h h;
#pragma unroll
        for (int i = 0; i < 8; ++i) h[i] = (_Float16)(oacc[mt][nt][i] * inv[i]);
        *(v8h*)(O + ((size_t)b * CDIM + c) * LDIM + q0 + mt * 16 + moff) = h;
      }
    }
  }
}

// ---------------------------------------------------------------------------
// Kernel 5: out-projection + bias + residual.
// ---------------------------------------------------------------------------
__global__ __launch_bounds__(256) void k_out_proj(
    const float* __restrict__ x, const _Float16* __restrict__ wout16,
    const float* __restrict__ bout, const _Float16* __restrict__ O,
    float* __restrict__ out) {
  __shared__ _Float16 Al[128 * 32];
  __shared__ _Float16 Bl[128 * 32];
  const int tid = threadIdx.x, lane = tid & 31, wid = tid >> 5;
  const int b = blockIdx.z;
  const int obase = blockIdx.y * 128;
  const int lbase = blockIdx.x * 128;
  const int wm = (wid >> 2) * 64;
  const int wn = (wid & 3) * 32;

  v8f acc[4][2];
#pragma unroll
  for (int m = 0; m < 4; ++m)
#pragma unroll
    for (int n = 0; n < 2; ++n) acc[m][n] = (v8f)0.f;

  for (int kk = 0; kk < CDIM; kk += 32) {
    {
      const int r = tid >> 1, ch = (tid & 1) * 16;
      *(v16h*)(Al + r * 32 + ch) =
          *(const v16h*)(wout16 + (size_t)(obase + r) * CDIM + kk + ch);
    }
    {
      const int c = tid >> 3, lch = (tid & 7) * 16;
      v16h ov = *(const v16h*)(O + ((size_t)b * CDIM + kk + c) * LDIM + lbase + lch);
#pragma unroll
      for (int j = 0; j < 16; ++j) Bl[(lch + j) * 32 + c] = ov[j];
    }
    __syncthreads();
    v16h af[4];
#pragma unroll
    for (int mt = 0; mt < 4; ++mt)
      af[mt] = load_a_frag(Al + (wm + mt * 16) * 32, 32, lane);
#pragma unroll
    for (int nt = 0; nt < 2; ++nt) {
      v16h bf = load_b_frag(Bl + (wn + nt * 16) * 32, 32, lane);
#pragma unroll
      for (int mt = 0; mt < 4; ++mt) acc[mt][nt] = wmma16(af[mt], bf, acc[mt][nt]);
    }
    __syncthreads();
  }

  const int lo = lane & 15;
  const int moff = (lane < 16) ? 0 : 8;
#pragma unroll
  for (int mt = 0; mt < 4; ++mt) {
    const int o0 = obase + wm + mt * 16 + moff;
#pragma unroll
    for (int nt = 0; nt < 2; ++nt) {
      const int l = lbase + wn + nt * 16 + lo;
      v8f t = acc[mt][nt];
#pragma unroll
      for (int i = 0; i < 8; ++i) {
        const size_t idx = ((size_t)b * CDIM + o0 + i) * LDIM + l;
        out[idx] = x[idx] + t[i] + bout[o0 + i];
      }
    }
  }
}

// ---------------------------------------------------------------------------
extern "C" void kernel_launch(void* const* d_in, const int* in_sizes, int n_in,
                              void* d_out, int out_size, void* d_ws, size_t ws_size,
                              hipStream_t stream) {
  (void)in_sizes; (void)n_in; (void)out_size; (void)ws_size;
  const float* x     = (const float*)d_in[0];
  const float* gamma = (const float*)d_in[1];
  const float* beta  = (const float*)d_in[2];
  const float* wqkv  = (const float*)d_in[3];
  const float* bqkv  = (const float*)d_in[4];
  const float* wout  = (const float*)d_in[5];
  const float* bout  = (const float*)d_in[6];
  float* out = (float*)d_out;

  char* ws = (char*)d_ws;
  size_t off = 0;
  auto alloc = [&](size_t bytes) -> char* {
    char* p = ws + off;
    off = (off + bytes + 255) & ~(size_t)255;
    return p;
  };
  _Float16* wqkv16 = (_Float16*)alloc((size_t)3 * CDIM * CDIM * 2);
  _Float16* wout16 = (_Float16*)alloc((size_t)CDIM * CDIM * 2);
  float*    scaleA = (float*)alloc((size_t)BATCH * CDIM * 4);
  float*    biasB  = (float*)alloc((size_t)BATCH * CDIM * 4);
  _Float16* qT     = (_Float16*)alloc((size_t)BATCH * LDIM * CDIM * 2);
  _Float16* kT     = (_Float16*)alloc((size_t)BATCH * LDIM * CDIM * 2);
  _Float16* vN     = (_Float16*)alloc((size_t)BATCH * CDIM * LDIM * 2);
  _Float16* Obuf   = (_Float16*)alloc((size_t)BATCH * CDIM * LDIM * 2);

  k_cvt_w<<<dim3((3 * CDIM * CDIM + 255) / 256), dim3(256), 0, stream>>>(
      wqkv, wout, wqkv16, wout16);
  k_gn_stats<<<dim3(BATCH * NGROUP), dim3(256), 0, stream>>>(
      x, gamma, beta, scaleA, biasB);
  k_gn_qkv_gemm<<<dim3(LDIM / 128, 3 * CDIM / 128, BATCH), dim3(256), 0, stream>>>(
      x, wqkv16, bqkv, scaleA, biasB, qT, kT, vN);
  // Qs 64K + Ks 128K + Vs 64K + Ss 16K + Ps 8K + row state
  const size_t shmem = 65536 + 131072 + 65536 + 16384 + 8192 + 3 * 64 * 4;
  k_flash<<<dim3(LDIM / 64, BATCH), dim3(256), shmem, stream>>>(qT, kT, vN, Obuf);
  k_out_proj<<<dim3(LDIM / 128, CDIM / 128, BATCH), dim3(256), 0, stream>>>(
      x, wout16, bout, Obuf, out);
}
